// MSARowAttentionWithPairBias_5600637354564
// MI455X (gfx1250) — compile-verified, matrix operand order checked by reading, not run on hardware
//
#include <hip/hip_runtime.h>
#include <hip/hip_bf16.h>
#include <math.h>

// MSARowAttentionWithPairBias for MI455X (gfx1250), wave32 + WMMA bf16.
// S=128, R=256, C_M=256, C_Z=128, H=8, C=32.

typedef __attribute__((ext_vector_type(16))) __bf16      v16bf;
typedef __attribute__((ext_vector_type(8)))  float       v8f;
typedef __attribute__((ext_vector_type(4)))  unsigned int u32x4;

#define NROW 32768  // S*R

// ---- WMMA fragment loaders (16x16x32 bf16, wave32 layouts from ISA 7.12.2) ----
// A (MxK, 16x32) and B (KxN via [N][K] memory): lane L<16 -> row L, K runs
// {0..7, 16..23}; lane L>=16 -> row L-16, K runs {8..15, 24..31}.
static __device__ __forceinline__ v16bf frag_ld_bf16(const __bf16* base, int ld, int lane) {
  int r  = lane & 15;
  int kb = (lane >> 4) << 3;
  const __bf16* p = base + r * ld + kb;
  union { u32x4 u[2]; v16bf v; } f;
  f.u[0] = *(const u32x4*)(p);        // K = kb .. kb+7
  f.u[1] = *(const u32x4*)(p + 16);   // K = kb+16 .. kb+23
  return f.v;
}

static __device__ __forceinline__ v16bf frag_cvt_f32(const float* base, int ld, int lane) {
  int r  = lane & 15;
  int kb = (lane >> 4) << 3;
  const float* p = base + r * ld + kb;
  v16bf v;
#pragma unroll
  for (int i = 0; i < 8; ++i) v[i] = (__bf16)p[i];
#pragma unroll
  for (int i = 0; i < 8; ++i) v[8 + i] = (__bf16)p[16 + i];
  return v;
}

#define WMMA_BF16(A, B, Cacc) \
  __builtin_amdgcn_wmma_f32_16x16x32_bf16(false, (A), false, (B), (short)0, (Cacc), false, false)

// ---- k0: fp32 weights -> bf16, transposed to [N][K] (WMMA-B friendly) ----
__global__ __launch_bounds__(256) void k0_wcvt(const float* __restrict__ wq,
                                               const float* __restrict__ wk,
                                               const float* __restrict__ wv,
                                               const float* __restrict__ wg,
                                               const float* __restrict__ wo,
                                               __bf16* __restrict__ wT) {
  int mat = blockIdx.x >> 8;     // 0..4
  int n   = blockIdx.x & 255;    // output col
  int k   = threadIdx.x;         // input  row
  const float* src = (mat == 0) ? wq : (mat == 1) ? wk : (mat == 2) ? wv
                   : (mat == 3) ? wg : wo;
  wT[(size_t)mat * 65536 + n * 256 + k] = (__bf16)src[k * 256 + n];
}

// ---- k1: LayerNorm(msa) + q/k/v/gate projections (WMMA) ----
// block = 256 thr (8 waves) handles 16 rows of [S*R, 256].
// wave w: projection p=w>>1, N-half (w&1)*8 .. +8 tiles; K=256 -> 8 wmma/tile.
__global__ __launch_bounds__(256) void k1_ln_qkvg(
    const float* __restrict__ msa, const float* __restrict__ lng,
    const float* __restrict__ lnb, const float* __restrict__ bg,
    const __bf16* __restrict__ wT,
    __bf16* __restrict__ qb, __bf16* __restrict__ kbuf,
    __bf16* __restrict__ vbuf, __bf16* __restrict__ gbuf) {
  __shared__ __align__(16) __bf16 ms[16][256];
  __shared__ float red[2][16][16];

  int t = threadIdx.x;
  int row0 = blockIdx.x * 16;
  int r = t >> 4, c16 = t & 15;
  const float* src = msa + (size_t)(row0 + r) * 256 + c16 * 16;
  float x[16]; float s = 0.f, s2 = 0.f;
#pragma unroll
  for (int i = 0; i < 16; ++i) { x[i] = src[i]; s += x[i]; s2 += x[i] * x[i]; }
  red[0][r][c16] = s; red[1][r][c16] = s2;
  __syncthreads();
  float mu = 0.f, m2 = 0.f;
#pragma unroll
  for (int i = 0; i < 16; ++i) { mu += red[0][r][i]; m2 += red[1][r][i]; }
  mu *= (1.f / 256.f);
  float var = m2 * (1.f / 256.f) - mu * mu;
  float rs = rsqrtf(var + 1e-5f);
#pragma unroll
  for (int i = 0; i < 16; ++i) {
    int cc = c16 * 16 + i;
    ms[r][cc] = (__bf16)((x[i] - mu) * rs * lng[cc] + lnb[cc]);
  }
  __syncthreads();

  int wave = t >> 5, lane = t & 31;
  int p   = wave >> 1;
  int nt0 = (wave & 1) * 8;
  const __bf16* W = wT + (size_t)p * 65536;

  v8f acc[8];
#pragma unroll
  for (int i = 0; i < 8; ++i) acc[i] = (v8f){0.f,0.f,0.f,0.f,0.f,0.f,0.f,0.f};

#pragma unroll
  for (int ks = 0; ks < 8; ++ks) {
    v16bf a = frag_ld_bf16(&ms[0][0] + ks * 32, 256, lane);
#pragma unroll
    for (int nt = 0; nt < 8; ++nt) {
      v16bf b = frag_ld_bf16(W + (nt0 + nt) * 16 * 256 + ks * 32, 256, lane);
      acc[nt] = WMMA_BF16(a, b, acc[nt]);
    }
  }

  __bf16* dst = (p == 0) ? qb : (p == 1) ? kbuf : (p == 2) ? vbuf : gbuf;
  int col = lane & 15, rb = (lane >> 4) * 8;
#pragma unroll
  for (int nt = 0; nt < 8; ++nt) {
    int ncol = (nt0 + nt) * 16 + col;
#pragma unroll
    for (int v = 0; v < 8; ++v) {
      float val = acc[nt][v];
      if (p == 0) val *= 0.17677669529663687f;                 // fold 1/sqrt(C) into q
      if (p == 3) val = 1.f / (1.f + __expf(-(val + bg[ncol])));  // gate = sigmoid(.+b_g)
      dst[(size_t)(row0 + rb + v) * 256 + ncol] = (__bf16)val;
    }
  }
}

// ---- k2: pair LayerNorm + [128 -> 8] head projection -> bias[H,R,R] ----
__global__ __launch_bounds__(256) void k2_pairbias(
    const float* __restrict__ pair, const float* __restrict__ lng,
    const float* __restrict__ lnb, const float* __restrict__ wb,
    const float* __restrict__ bb, float* __restrict__ bias) {
  int t = threadIdx.x;
  int wave = t >> 5, lane = t & 31;
  int row = blockIdx.x * 8 + wave;            // i*256+j in [0,65536)
  const float* zr = pair + (size_t)row * 128;
  int c0 = lane * 4;
  float x[4]; float s = 0.f, s2 = 0.f;
#pragma unroll
  for (int i = 0; i < 4; ++i) { x[i] = zr[c0 + i]; s += x[i]; s2 += x[i] * x[i]; }
#pragma unroll
  for (int m = 16; m >= 1; m >>= 1) { s += __shfl_xor(s, m, 32); s2 += __shfl_xor(s2, m, 32); }
  float mu = s * (1.f / 128.f);
  float var = s2 * (1.f / 128.f) - mu * mu;
  float rs = rsqrtf(var + 1e-5f);
  float acc[8];
#pragma unroll
  for (int h = 0; h < 8; ++h) acc[h] = 0.f;
#pragma unroll
  for (int i = 0; i < 4; ++i) {
    int c = c0 + i;
    float zn = (x[i] - mu) * rs * lng[c] + lnb[c];
#pragma unroll
    for (int h = 0; h < 8; ++h) acc[h] += zn * wb[c * 8 + h];
  }
#pragma unroll
  for (int m = 16; m >= 1; m >>= 1) {
#pragma unroll
    for (int h = 0; h < 8; ++h) acc[h] += __shfl_xor(acc[h], m, 32);
  }
  if (lane == 0) {
#pragma unroll
    for (int h = 0; h < 8; ++h) bias[(size_t)h * 65536 + row] = acc[h] + bb[h];
  }
}

// ---- k3: fused attention per (s, h, 32-query tile): QK^T+bias, softmax, PV, gate ----
// LDS: vts 16KB + Sb 32KB + reductions 2KB = 50KB (<< 320KB/WGP).
__global__ __launch_bounds__(256) void k3_attn(
    const __bf16* __restrict__ qb, const __bf16* __restrict__ kbuf,
    const __bf16* __restrict__ vbuf, const __bf16* __restrict__ gbuf,
    const float* __restrict__ bias, __bf16* __restrict__ ob) {
  __shared__ __align__(16) __bf16 vts[32][256];   // v transposed: [ch][key]
  __shared__ __align__(16) float  Sb[32][256];    // logits -> probs
  __shared__ float rmax[32][8];
  __shared__ float rsum[32][8];

  int t   = threadIdx.x;
  int bid = blockIdx.x;
  int qt = bid & 7;          // query tile (R/32 = 8)
  int h  = (bid >> 3) & 7;   // head
  int s  = bid >> 6;         // MSA row
  int q0 = qt * 32;

  const __bf16* qp = qb   + (size_t)(s * 256 + q0) * 256 + h * 32;
  const __bf16* kp = kbuf + (size_t)s * 256 * 256 + h * 32;
  const __bf16* vp = vbuf + (size_t)s * 256 * 256 + h * 32;

  // stage v transposed into LDS: thread t reads its key row (64B) as 4x b128,
  // then does the unavoidable 2-byte transpose scatter into vts[ch][key].
  {
    const u32x4* vr4 = (const u32x4*)(vp + (size_t)t * 256);
    union { u32x4 u[4]; __bf16 hln[32]; } vv;
#pragma unroll
    for (int i = 0; i < 4; ++i) vv.u[i] = vr4[i];
#pragma unroll
    for (int ch = 0; ch < 32; ++ch) vts[ch][t] = vv.hln[ch];
  }

  int wave = t >> 5, lane = t & 31;

  // logits: 2 M-tiles x 16 N-tiles = 32 wmma tiles, 4 per wave, K=32 each
#pragma unroll
  for (int i = 0; i < 4; ++i) {
    int tt = wave * 4 + i;
    int mt = tt >> 4, ntg = tt & 15;
    v16bf a = frag_ld_bf16(qp + mt * 16 * 256, 256, lane);
    v16bf b = frag_ld_bf16(kp + ntg * 16 * 256, 256, lane);
    v8f accS = (v8f){0.f,0.f,0.f,0.f,0.f,0.f,0.f,0.f};
    accS = WMMA_BF16(a, b, accS);
    int col = lane & 15, rb = (lane >> 4) * 8;
    const float* bp = bias + (size_t)h * 65536 + (size_t)(q0 + mt * 16) * 256 + ntg * 16 + col;
#pragma unroll
    for (int v = 0; v < 8; ++v)
      Sb[mt * 16 + rb + v][ntg * 16 + col] = accS[v] + bp[(rb + v) * 256];
  }
  __syncthreads();

  // softmax over 256 keys; 8 threads per row, 32 elems each
  int row = t >> 3, seg = t & 7;
  float* srow = &Sb[row][seg * 32];
  float mx = -3.0e38f;
#pragma unroll
  for (int j = 0; j < 32; ++j) mx = fmaxf(mx, srow[j]);
  rmax[row][seg] = mx;
  __syncthreads();
  float M = -3.0e38f;
#pragma unroll
  for (int i = 0; i < 8; ++i) M = fmaxf(M, rmax[row][i]);
  float sum = 0.f;
#pragma unroll
  for (int j = 0; j < 32; ++j) { float e = __expf(srow[j] - M); srow[j] = e; sum += e; }
  rsum[row][seg] = sum;
  __syncthreads();
  float tot = 0.f;
#pragma unroll
  for (int i = 0; i < 8; ++i) tot += rsum[row][i];
  float inv = 1.f / tot;
#pragma unroll
  for (int j = 0; j < 32; ++j) srow[j] *= inv;
  __syncthreads();

  // O = P * V : 2 M-tiles x 2 N-tiles, K=256 -> 8 wmma per tile; waves 0-3
  if (wave < 4) {
    int mt = wave >> 1, ntv = wave & 1;
    v8f acc = (v8f){0.f,0.f,0.f,0.f,0.f,0.f,0.f,0.f};
#pragma unroll
    for (int ks = 0; ks < 8; ++ks) {
      v16bf a = frag_cvt_f32(&Sb[mt * 16][0] + ks * 32, 256, lane);
      v16bf b = frag_ld_bf16(&vts[ntv * 16][0] + ks * 32, 256, lane);
      acc = WMMA_BF16(a, b, acc);
    }
    int col = lane & 15, rb = (lane >> 4) * 8, ch = ntv * 16 + col;
#pragma unroll
    for (int v = 0; v < 8; ++v) {
      int qr = q0 + mt * 16 + rb + v;
      size_t idx = (size_t)(s * 256 + qr) * 256 + h * 32 + ch;
      float g = (float)gbuf[idx];
      ob[idx] = (__bf16)(acc[v] * g);
    }
  }
}

// ---- k4: output projection [32768,256] x woT[256,256] + b_o -> f32 out ----
// Tile copy global->LDS uses CDNA5 async DMA (GLOBAL_LOAD_ASYNC_TO_LDS_B128,
// ASYNCcnt-tracked); INST_OFFSET is added to both LDS and global addresses.
__global__ __launch_bounds__(256) void k4_outproj(
    const __bf16* __restrict__ ob, const __bf16* __restrict__ woT,
    const float* __restrict__ bo, float* __restrict__ out) {
  __shared__ __align__(16) __bf16 os[16][256];
  int t = threadIdx.x;
  int row0 = blockIdx.x * 16;
  int r = t >> 4, c16 = t & 15;
  {
    unsigned laddr = (unsigned)(size_t)&os[r][c16 * 16];   // low 32 bits = LDS offset
    unsigned long long gaddr =
        (unsigned long long)(const void*)(ob + (size_t)(row0 + r) * 256 + c16 * 16);
    asm volatile(
        "global_load_async_to_lds_b128 %0, %1, off\n\t"
        "global_load_async_to_lds_b128 %0, %1, off offset:16"
        :: "v"(laddr), "v"(gaddr) : "memory");
  }
  asm volatile("s_wait_asynccnt 0" ::: "memory");
  __syncthreads();

  int wave = t >> 5, lane = t & 31;
  v8f acc[2];
  acc[0] = (v8f){0.f,0.f,0.f,0.f,0.f,0.f,0.f,0.f};
  acc[1] = (v8f){0.f,0.f,0.f,0.f,0.f,0.f,0.f,0.f};
#pragma unroll
  for (int ks = 0; ks < 8; ++ks) {
    v16bf a = frag_ld_bf16(&os[0][0] + ks * 32, 256, lane);
#pragma unroll
    for (int nt = 0; nt < 2; ++nt) {
      v16bf b = frag_ld_bf16(woT + (wave * 2 + nt) * 16 * 256 + ks * 32, 256, lane);
      acc[nt] = WMMA_BF16(a, b, acc[nt]);
    }
  }
  int col = lane & 15, rb = (lane >> 4) * 8;
#pragma unroll
  for (int nt = 0; nt < 2; ++nt) {
    int ncol = (wave * 2 + nt) * 16 + col;
#pragma unroll
    for (int v = 0; v < 8; ++v)
      out[(size_t)(row0 + rb + v) * 256 + ncol] = acc[nt][v] + bo[ncol];
  }
}

extern "C" void kernel_launch(void* const* d_in, const int* in_sizes, int n_in,
                              void* d_out, int out_size, void* d_ws, size_t ws_size,
                              hipStream_t stream) {
  (void)in_sizes; (void)n_in; (void)out_size; (void)ws_size;
  const float* msa  = (const float*)d_in[0];
  const float* pair = (const float*)d_in[1];
  const float* lnmg = (const float*)d_in[2];
  const float* lnmb = (const float*)d_in[3];
  const float* lnpg = (const float*)d_in[4];
  const float* lnpb = (const float*)d_in[5];
  const float* wq   = (const float*)d_in[6];
  const float* wk   = (const float*)d_in[7];
  const float* wv   = (const float*)d_in[8];
  const float* wg   = (const float*)d_in[9];
  const float* bg   = (const float*)d_in[10];
  const float* wb   = (const float*)d_in[11];
  const float* bb   = (const float*)d_in[12];
  const float* wo   = (const float*)d_in[13];
  const float* bo   = (const float*)d_in[14];
  float* out = (float*)d_out;

  char* ws = (char*)d_ws;
  size_t off = 0;
  const size_t ROWB = (size_t)NROW * 256;          // 8,388,608 elems
  __bf16* qb = (__bf16*)(ws + off); off += ROWB * 2;
  __bf16* kb = (__bf16*)(ws + off); off += ROWB * 2;
  __bf16* vb = (__bf16*)(ws + off); off += ROWB * 2;
  __bf16* gb = (__bf16*)(ws + off); off += ROWB * 2;
  __bf16* ob = (__bf16*)(ws + off); off += ROWB * 2;
  float*  bias = (float*)(ws + off); off += (size_t)8 * 65536 * 4;
  __bf16* wT = (__bf16*)(ws + off); off += (size_t)5 * 65536 * 2;

  k0_wcvt   <<<1280, 256, 0, stream>>>(wq, wk, wv, wg, wo, wT);
  k1_ln_qkvg<<<2048, 256, 0, stream>>>(msa, lnmg, lnmb, bg, wT, qb, kb, vb, gb);
  k2_pairbias<<<8192, 256, 0, stream>>>(pair, lnpg, lnpb, wb, bb, bias);
  k3_attn   <<<8192, 256, 0, stream>>>(qb, kb, vb, gb, bias, ob);
  k4_outproj<<<2048, 256, 0, stream>>>(ob, wT + (size_t)4 * 65536, bo, out);
}
